// PASScell_1_12962211300015
// MI455X (gfx1250) — matrix operand order repaired and checked by reading, not compiled, not run on hardware
//
#include <hip/hip_runtime.h>

#define IN_DIM 256
#define HID    1024
#define BATCH  16384
#define KTOT   (IN_DIM + HID)          // 1280

#define BM  128   // rows per workgroup
#define BN  64    // cols per gate per workgroup
#define BK  32    // K per step
#define BKP 40    // padded LDS K stride (80B rows -> conflict-free b128 frag loads)
#define NG  4     // gates

#define AS_BYTES  (BM * BKP * 2)             // 10240
#define BS_BYTES  (NG * BN * BKP * 2)        // 20480
#define BUF_BYTES (AS_BYTES + BS_BYTES)      // 30720, x2 for double buffering

#define WS_AB_BYTES ((size_t)BATCH * KTOT * 2)      // 40 MB: [x|h] bf16 row-major
#define WS_BT_BYTES ((size_t)NG * HID * KTOT * 2)   // 10 MB: weights bf16, N-major
#define WS_NEEDED   (WS_AB_BYTES + WS_BT_BYTES)

typedef __attribute__((ext_vector_type(16))) __bf16       v16bf;
typedef __attribute__((ext_vector_type(8)))  __bf16       v8bf;
typedef __attribute__((ext_vector_type(4)))  __bf16       v4bf;
typedef __attribute__((ext_vector_type(8)))  float        v8f;
typedef __attribute__((ext_vector_type(4)))  unsigned int v4u;
typedef __attribute__((ext_vector_type(8)))  int          v8i;
typedef __attribute__((ext_vector_type(4)))  int          v4i;

__device__ __forceinline__ float sigmoidf_fast(float v) {
    return 1.0f / (1.0f + __expf(-v));
}

// ---------------------------------------------------------------------------
// TDM: one tensor_load_to_lds issue. Tile = 32(k) x tile_d1 x tile_d2, bf16,
// innermost dim contiguous in memory (stride KTOT rows). pad_enable inserts
// 16B of LDS padding after every 64B row -> 80B LDS row stride (== BKP).
// D# layout per CDNA5 ISA ch.8. 6-arg builtin: 5th (int32x8) operand is the
// unused/NULL descriptor slot, zero-filled; last arg = cache policy.
// ---------------------------------------------------------------------------
__device__ __forceinline__ void tdm_load_tile(unsigned lds_off, const void* gptr,
                                              unsigned tensor_d1, unsigned tile_d1,
                                              unsigned tile_d2,
                                              unsigned long long zstride_elems)
{
    unsigned long long ga = (unsigned long long)(uintptr_t)gptr;
    v4u g0 = { 1u,                                    // count=1, is_restore=0
               lds_off,                               // lds byte address
               (unsigned)ga,                          // global_addr[31:0]
               (unsigned)((ga >> 32) & 0x01FFFFFFull) // global_addr[56:32]
                   | 0x80000000u };                   // type=2 ("image")
    v8i g1 = {
        (int)((1u << 16)      // data_size = 1 -> 2 bytes
            | (1u << 20)      // pad_enable
            | (3u << 22)      // pad_interval: 16 DWORDs (64B) between pads
            | (3u << 25)),    // pad_amount: 4 DWORDs (16B)
        (int)(((unsigned)KTOT & 0xFFFFu) << 16),                  // tensor_dim0 lo
        (int)(((unsigned)KTOT >> 16) | ((tensor_d1 & 0xFFFFu) << 16)),
        (int)((tensor_d1 >> 16) | (32u << 16)),                   // tile_dim0 = 32
        (int)(tile_d1 | (tile_d2 << 16)),                         // tile_dim1/2
        (int)KTOT,                                                // dim0_stride lo32
        (int)((unsigned)(zstride_elems & 0xFFFFull) << 16),       // dim1_stride lo16
        (int)(zstride_elems >> 16)                                // dim1_stride hi32
    };
    v4i g2 = { (int)((tile_d2 == 0) ? 1u : tile_d2), 0, 0, 0 };   // tensor_dim2
    v4i g3 = { 0, 0, 0, 0 };
    v8i g4 = { 0, 0, 0, 0, 0, 0, 0, 0 };                          // unused slot
    __builtin_amdgcn_tensor_load_to_lds(g0, g1, g2, g3, g4, 0);
}

// ---------------------------------------------------------------------------
// Pre-pass 1: Ab[row][k] bf16, k<256 from x, else from h   (coalesced f32->bf16)
// ---------------------------------------------------------------------------
__global__ __launch_bounds__(256) void convert_A_kernel(const float* __restrict__ x,
                                                        const float* __restrict__ h,
                                                        __bf16* __restrict__ Ab)
{
    const size_t idx = (size_t)blockIdx.x * blockDim.x + threadIdx.x;
    const size_t e   = idx * 4;                     // 4 elems per thread
    const int row = (int)(e / KTOT);
    const int k   = (int)(e % KTOT);
    const float4 v = (k < IN_DIM)
        ? *(const float4*)(x + (size_t)row * IN_DIM + k)
        : *(const float4*)(h + (size_t)row * HID + (k - IN_DIM));
    v4bf b4 = { (__bf16)v.x, (__bf16)v.y, (__bf16)v.z, (__bf16)v.w };
    *(v4bf*)(Ab + e) = b4;
}

// ---------------------------------------------------------------------------
// Pre-pass 2: Bt[col][k] bf16 (N-major / transposed), k<256 from W, else from R
// One gate per launch.
// ---------------------------------------------------------------------------
__global__ __launch_bounds__(256) void convert_B_kernel(const float* __restrict__ W,
                                                        const float* __restrict__ R,
                                                        __bf16* __restrict__ Bt)
{
    const size_t idx = (size_t)blockIdx.x * blockDim.x + threadIdx.x;
    const int col = (int)(idx / (KTOT / 4));
    const int k4  = (int)(idx % (KTOT / 4)) * 4;    // 4 consecutive k, never straddle 256
    const float* src = (k4 < IN_DIM)
        ? (W + (size_t)k4 * HID + col)
        : (R + (size_t)(k4 - IN_DIM) * HID + col);
    const float a0 = src[0];
    const float a1 = src[HID];
    const float a2 = src[2 * HID];
    const float a3 = src[3 * HID];
    v4bf b4 = { (__bf16)a0, (__bf16)a1, (__bf16)a2, (__bf16)a3 };
    *(v4bf*)(Bt + (size_t)col * KTOT + k4) = b4;
}

// ---------------------------------------------------------------------------
// Main GEMM: bf16 WMMA, TDM-staged LDS, double buffered, fused gate epilogue.
// Wave blocking: 2 M-subtiles x (4 gates x 2 N-subtiles) per wave
//   -> 16 WMMAs per 10 fragment loads per K-step.
// ---------------------------------------------------------------------------
__global__ __launch_bounds__(256) void
plstm_gemm_tdm_kernel(const __bf16* __restrict__ Ab,
                      const __bf16* __restrict__ Bt,
                      const float* __restrict__ o,
                      const float* __restrict__ bf, const float* __restrict__ bi,
                      const float* __restrict__ bk, const float* __restrict__ bg,
                      float* __restrict__ out)
{
    extern __shared__ __align__(16) char smem[];   // 2 * BUF_BYTES

    const int t    = threadIdx.x;
    const int lane = t & 31;
    const int wv   = t >> 5;        // wave id 0..7
    const int half = lane >> 4;
    const int p    = lane & 15;

    const int mp = wv & 3;          // M-pair: 2 subtiles of 16 rows each
    const int np = wv >> 2;         // N-pair: 2 subtiles of 16 cols each

    const int rowBase = blockIdx.y * BM;
    const int colBase = blockIdx.x * BN;

    v8f acc[NG][2][2];              // [gate][n-sub][m-sub]
    #pragma unroll
    for (int g = 0; g < NG; ++g)
        #pragma unroll
        for (int j = 0; j < 2; ++j)
            #pragma unroll
            for (int i = 0; i < 2; ++i)
                acc[g][j][i] = (v8f){0.f, 0.f, 0.f, 0.f, 0.f, 0.f, 0.f, 0.f};

    // issue one K-step's DMA: A tile (128x32) + all 4 gate B tiles (4x64x32, 3D)
    auto issue_step = [&](int s, int buf) {
        const int kk = s * BK;
        tdm_load_tile((unsigned)(buf * BUF_BYTES),
                      Ab + (size_t)rowBase * KTOT + kk,
                      (unsigned)BATCH, (unsigned)BM, 0u, 0ull);
        tdm_load_tile((unsigned)(buf * BUF_BYTES + AS_BYTES),
                      Bt + (size_t)colBase * KTOT + kk,
                      (unsigned)HID, (unsigned)BN, (unsigned)NG,
                      (unsigned long long)HID * KTOT);
    };

    // 16 WMMAs on buffer `cur`: each B fragment feeds 2 WMMAs (m0, m1)
    auto compute_step = [&](const int cur) {
        const __bf16* AsP = (const __bf16*)(smem + cur * BUF_BYTES);
        const __bf16* BsP = (const __bf16*)(smem + cur * BUF_BYTES + AS_BYTES);

        const int ka = half << 3;     // A: lanes0-15 K{0-7,16-23}, 16-31 K{8-15,24-31}
        const int kb = half << 3;     // B: lanes0-15 K{0-7,16-23}, 16-31 K{8-15,24-31}

        v16bf af[2];
        #pragma unroll
        for (int i = 0; i < 2; ++i) {
            const int arow = mp * 32 + i * 16 + p;
            v8bf lo = *(const v8bf*)(AsP + arow * BKP + ka);
            v8bf hi = *(const v8bf*)(AsP + arow * BKP + ka + 16);
            af[i] = __builtin_shufflevector(lo, hi,
                        0,1,2,3,4,5,6,7,8,9,10,11,12,13,14,15);
        }

        #pragma unroll
        for (int g = 0; g < NG; ++g) {
            #pragma unroll
            for (int j = 0; j < 2; ++j) {
                const int brow = g * BN + (np * 2 + j) * 16 + p;
                v8bf lo = *(const v8bf*)(BsP + brow * BKP + kb);
                v8bf hi = *(const v8bf*)(BsP + brow * BKP + kb + 16);
                v16bf bfm = __builtin_shufflevector(lo, hi,
                                0,1,2,3,4,5,6,7,8,9,10,11,12,13,14,15);
                acc[g][j][0] = __builtin_amdgcn_wmma_f32_16x16x32_bf16(
                    false, af[0], false, bfm, (short)0, acc[g][j][0], false, false);
                acc[g][j][1] = __builtin_amdgcn_wmma_f32_16x16x32_bf16(
                    false, af[1], false, bfm, (short)0, acc[g][j][1], false, false);
            }
        }
    };

    const int NSTEPS = KTOT / BK;   // 40 (even)
    if (wv == 0) issue_step(0, 0);

    for (int s = 0; s < NSTEPS; s += 2) {
        // ---- half A: compute buf0, prefetch s+1 -> buf1 (s+1 < NSTEPS always)
        if (wv == 0) {
            issue_step(s + 1, 1);
            __builtin_amdgcn_s_wait_tensorcnt(2);   // step s's 2 TDMs done
        }
        __syncthreads();
        compute_step(0);
        __syncthreads();
        // ---- half B: compute buf1, prefetch s+2 -> buf0
        if (wv == 0) {
            if (s + 2 < NSTEPS) {
                issue_step(s + 2, 0);
                __builtin_amdgcn_s_wait_tensorcnt(2);
            } else {
                __builtin_amdgcn_s_wait_tensorcnt(0);
            }
        }
        __syncthreads();
        compute_step(1);
        __syncthreads();
    }

    // fused epilogue: sigmoid gates -> o_new, h_new (non-temporal stream I/O)
    const size_t OUT2 = (size_t)BATCH * HID;
    #pragma unroll
    for (int j = 0; j < 2; ++j) {
        const int col = colBase + (np * 2 + j) * 16 + p;
        const float bfv = bf[col];
        const float biv = bi[col];
        const float bkv = bk[col];
        const float bgv = bg[col];
        #pragma unroll
        for (int i = 0; i < 2; ++i) {
            #pragma unroll
            for (int v = 0; v < 8; ++v) {
                const int row = rowBase + mp * 32 + i * 16 + (half << 3) + v;  // C/D layout
                const float fv = sigmoidf_fast(acc[0][j][i][v] + bfv);
                const float iv = sigmoidf_fast(acc[1][j][i][v] + biv);
                const float kv = sigmoidf_fast(acc[2][j][i][v] + bkv);
                const float gv = sigmoidf_fast(acc[3][j][i][v] + bgv);
                const float oo = __builtin_nontemporal_load(o + (size_t)row * HID + col);
                const float on = fv * oo - iv * fmaxf(gv, 0.0f);
                const float hn = kv * fmaxf(on, 0.0f);
                __builtin_nontemporal_store(on, out + (size_t)row * HID + col);
                __builtin_nontemporal_store(hn, out + OUT2 + (size_t)row * HID + col);
            }
        }
    }
}

// ---------------------------------------------------------------------------
// Fallback (no workspace): round-1 kernel, f32->bf16 conversion in-kernel.
// ---------------------------------------------------------------------------
__global__ __launch_bounds__(256) void
plstm_cell_wmma_fallback(const float* __restrict__ x,
                         const float* __restrict__ o,
                         const float* __restrict__ h,
                         const float* __restrict__ Wf, const float* __restrict__ Wi,
                         const float* __restrict__ Wk, const float* __restrict__ Wg,
                         const float* __restrict__ Rf, const float* __restrict__ Ri,
                         const float* __restrict__ Rk, const float* __restrict__ Rg,
                         const float* __restrict__ bf, const float* __restrict__ bi,
                         const float* __restrict__ bk, const float* __restrict__ bg,
                         float* __restrict__ out)
{
    __shared__ __bf16 As[BM][BKP];
    __shared__ __bf16 Bs[NG][BN][BKP];

    const int t    = threadIdx.x;
    const int lane = t & 31;
    const int wv   = t >> 5;
    const int half = lane >> 4;
    const int p    = lane & 15;

    const int rowBase = blockIdx.y * BM;
    const int colBase = blockIdx.x * BN;

    v8f acc[NG][4];
    #pragma unroll
    for (int g = 0; g < NG; ++g)
        #pragma unroll
        for (int n = 0; n < 4; ++n)
            acc[g][n] = (v8f){0.f, 0.f, 0.f, 0.f, 0.f, 0.f, 0.f, 0.f};

    auto do_step = [&](const float* __restrict__ aBase, int lda,
                       const float* __restrict__ w0, const float* __restrict__ w1,
                       const float* __restrict__ w2, const float* __restrict__ w3,
                       int kk) {
        #pragma unroll
        for (int i = 0; i < 4; ++i) {
            int slot = t + i * 256;
            int row  = slot >> 3;
            int kq   = (slot & 7) << 2;
            const float4 v = *(const float4*)(aBase + (size_t)(rowBase + row) * lda + kk + kq);
            v4bf b4 = { (__bf16)v.x, (__bf16)v.y, (__bf16)v.z, (__bf16)v.w };
            *(v4bf*)&As[row][kq] = b4;
        }
        const float* wp[NG] = { w0, w1, w2, w3 };
        #pragma unroll
        for (int i = 0; i < 8; ++i) {
            int g   = i >> 1;
            int rem = ((i & 1) << 8) + t;
            int k   = rem >> 4;
            int nq  = (rem & 15) << 2;
            const float4 v = *(const float4*)(wp[g] + (size_t)(kk + k) * HID + colBase + nq);
            Bs[g][nq + 0][k] = (__bf16)v.x;
            Bs[g][nq + 1][k] = (__bf16)v.y;
            Bs[g][nq + 2][k] = (__bf16)v.z;
            Bs[g][nq + 3][k] = (__bf16)v.w;
        }
        __syncthreads();

        const int arow = wv * 16 + p;
        const int ka   = half << 3;
        v8bf alo = *(const v8bf*)&As[arow][ka];
        v8bf ahi = *(const v8bf*)&As[arow][ka + 16];
        v16bf af = __builtin_shufflevector(alo, ahi,
                       0,1,2,3,4,5,6,7,8,9,10,11,12,13,14,15);

        #pragma unroll
        for (int g = 0; g < NG; ++g) {
            #pragma unroll
            for (int n = 0; n < 4; ++n) {
                const int brow = n * 16 + p;
                const int kb   = half << 3;
                v8bf blo = *(const v8bf*)&Bs[g][brow][kb];
                v8bf bhi = *(const v8bf*)&Bs[g][brow][kb + 16];
                v16bf bfm = __builtin_shufflevector(blo, bhi,
                               0,1,2,3,4,5,6,7,8,9,10,11,12,13,14,15);
                acc[g][n] = __builtin_amdgcn_wmma_f32_16x16x32_bf16(
                                false, af, false, bfm, (short)0, acc[g][n],
                                false, false);
            }
        }
        __syncthreads();
    };

    for (int kk = 0; kk < IN_DIM; kk += BK)
        do_step(x, IN_DIM, Wf, Wi, Wk, Wg, kk);
    for (int kk = 0; kk < HID; kk += BK)
        do_step(h, HID, Rf, Ri, Rk, Rg, kk);

    const size_t OUT2 = (size_t)BATCH * HID;
    #pragma unroll
    for (int n = 0; n < 4; ++n) {
        const int col = colBase + n * 16 + p;
        const float bfv = bf[col];
        const float biv = bi[col];
        const float bkv = bk[col];
        const float bgv = bg[col];
        #pragma unroll
        for (int v = 0; v < 8; ++v) {
            const int row = rowBase + wv * 16 + (half << 3) + v;
            const float fv = sigmoidf_fast(acc[0][n][v] + bfv);
            const float iv = sigmoidf_fast(acc[1][n][v] + biv);
            const float kv = sigmoidf_fast(acc[2][n][v] + bkv);
            const float gv = sigmoidf_fast(acc[3][n][v] + bgv);
            const float oo = __builtin_nontemporal_load(o + (size_t)row * HID + col);
            const float on = fv * oo - iv * fmaxf(gv, 0.0f);
            const float hn = kv * fmaxf(on, 0.0f);
            __builtin_nontemporal_store(on, out + (size_t)row * HID + col);
            __builtin_nontemporal_store(hn, out + OUT2 + (size_t)row * HID + col);
        }
    }
}

extern "C" void kernel_launch(void* const* d_in, const int* in_sizes, int n_in,
                              void* d_out, int out_size, void* d_ws, size_t ws_size,
                              hipStream_t stream) {
    (void)in_sizes; (void)n_in; (void)out_size;
    const float* x   = (const float*)d_in[0];
    const float* o   = (const float*)d_in[1];
    const float* h   = (const float*)d_in[2];
    const float* W_f = (const float*)d_in[3];
    const float* b_f = (const float*)d_in[4];
    const float* R_f = (const float*)d_in[5];
    const float* W_i = (const float*)d_in[6];
    const float* b_i = (const float*)d_in[7];
    const float* R_i = (const float*)d_in[8];
    const float* W_k = (const float*)d_in[9];
    const float* b_k = (const float*)d_in[10];
    const float* R_k = (const float*)d_in[11];
    const float* W_g = (const float*)d_in[12];
    const float* b_g = (const float*)d_in[13];
    const float* R_g = (const float*)d_in[14];
    float* out = (float*)d_out;

    dim3 grid(HID / BN, BATCH / BM);   // (16, 128)
    dim3 block(256);                   // 8 waves (wave32)

    if (ws_size >= WS_NEEDED) {
        __bf16* Ab = (__bf16*)d_ws;
        __bf16* Bt = (__bf16*)((char*)d_ws + WS_AB_BYTES);

        // pre-pass conversions (bandwidth-trivial: ~150MB total)
        convert_A_kernel<<<(unsigned)((size_t)BATCH * KTOT / 4 / 256), 256, 0, stream>>>(x, h, Ab);
        const float* Ws[NG] = { W_f, W_i, W_k, W_g };
        const float* Rs[NG] = { R_f, R_i, R_k, R_g };
        for (int g = 0; g < NG; ++g)
            convert_B_kernel<<<(unsigned)((size_t)HID * (KTOT / 4) / 256), 256, 0, stream>>>(
                Ws[g], Rs[g], Bt + (size_t)g * HID * KTOT);

        plstm_gemm_tdm_kernel<<<grid, block, 2 * BUF_BYTES, stream>>>(
            Ab, Bt, o, b_f, b_i, b_k, b_g, out);
    } else {
        plstm_cell_wmma_fallback<<<grid, block, 0, stream>>>(
            x, o, h,
            W_f, W_i, W_k, W_g,
            R_f, R_i, R_k, R_g,
            b_f, b_i, b_k, b_g,
            out);
    }
}
